// ParallelExperts_3126736191877
// MI455X (gfx1250) — matrix operand order repaired and checked by reading, not compiled, not run on hardware
//
#include <hip/hip_runtime.h>
#include <math.h>

// ---------------------------------------------------------------------------
// Problem constants (reference setup): B=4,S=4096,H=1024,D=1024,E=64,CAP=2048,
// T = B*S = 16384.
// ---------------------------------------------------------------------------
#define T_TOK   16384
#define HDIM    1024
#define DDIM    1024
#define NEXP    64
#define CAP     2048

typedef __attribute__((ext_vector_type(16))) __bf16 v16bf;
typedef __attribute__((ext_vector_type(8)))  __bf16 v8bf;
typedef __attribute__((ext_vector_type(8)))  float  v8f;
typedef unsigned __attribute__((ext_vector_type(4))) u32x4;
typedef int      __attribute__((ext_vector_type(4))) i32x4;
typedef int      __attribute__((ext_vector_type(8))) i32x8;

union V16U {
    v16bf v;
    v8bf  h[2];
};

__device__ __forceinline__ unsigned short f2bf(float f) {
    unsigned u = __float_as_uint(f);
    u += 0x7FFFu + ((u >> 16) & 1u);   // round-to-nearest-even
    return (unsigned short)(u >> 16);
}

// Async global -> LDS copy of 16 bytes per lane (ASYNCcnt-tracked).
#define ASYNC_CP16(ldsPtr, gPtr)                                             \
    asm volatile("global_load_async_to_lds_b128 %0, %1, off"                 \
                 :: "v"((unsigned)(uintptr_t)(ldsPtr)),                      \
                    "v"((unsigned long long)(uintptr_t)(gPtr))               \
                 : "memory")

#define WAIT_ASYNC0() asm volatile("s_wait_asynccnt 0x0" ::: "memory")

// ---------------------------------------------------------------------------
// TDM: load a 2D bf16 tile (tile_dim0=32 elems contiguous, tile_dim1=128 rows,
// row stride 1024 elems) from global into LDS, inserting 16 B of LDS padding
// after every 64 B row (-> 80 B row stride == LDT*2). TENSORcnt-tracked.
// Descriptor packing per CDNA5 ISA ch.8 (D# groups 0/1; 2-D tensor so groups
// 2/3 are zero/disabled).
// ---------------------------------------------------------------------------
__device__ __forceinline__ void tdm_load_tile_bf16(
    unsigned ldsAddr, const unsigned short* gTile)
{
    const unsigned long long ga = (unsigned long long)(uintptr_t)gTile;
    u32x4 g0;
    g0.x = 1u;                                   // count=1, user mode
    g0.y = ldsAddr;                              // lds_addr (bytes)
    g0.z = (unsigned)ga;                         // global_addr[31:0]
    g0.w = (unsigned)((ga >> 32) & 0x01FFFFFFu)  // global_addr[56:32]
         | 0x80000000u;                          // type=2 ("image")
    i32x8 g1;
    g1[0] = (int)((1u << 16)       // data_size = 1 -> 2 bytes
                | (1u << 20)       // pad_enable
                | (3u << 22)       // pad_interval: 16 DWORDs (64 B)
                | (3u << 25));     // pad_amount: 4 DWORDs (16 B)
    g1[1] = (int)(1024u << 16);    // tensor_dim0[15:0] at bit48
    g1[2] = (int)(1024u << 16);    // tensor_dim0[31:16]=0 | tensor_dim1[15:0]
    g1[3] = (int)(32u << 16);      // tensor_dim1[31:16]=0 | tile_dim0=32
    g1[4] = 128;                   // tile_dim1=128 | tile_dim2=0
    g1[5] = 1024;                  // tensor_dim0_stride[31:0] (elements)
    g1[6] = 0;                     // stride[47:32]=0 | tensor_dim1_stride lo
    g1[7] = 0;
    const i32x4 z4 = {0, 0, 0, 0};
#if defined(__clang_major__) && (__clang_major__ >= 23)
    const i32x8 z8 = {0, 0, 0, 0, 0, 0, 0, 0};
    __builtin_amdgcn_tensor_load_to_lds(g0, g1, z4, z4, z8, 0);
#else
    __builtin_amdgcn_tensor_load_to_lds(g0, g1, z4, z4, 0);
#endif
}

// ---------------------------------------------------------------------------
// Elementwise fp32 -> bf16 (RNE), 8 elements/thread.
// ---------------------------------------------------------------------------
__global__ __launch_bounds__(256) void cvt_bf16_kernel(
    const float* __restrict__ src, unsigned short* __restrict__ dst)
{
    const size_t i = ((size_t)blockIdx.x * 256 + threadIdx.x) * 8;
    const float4 a = *(const float4*)(src + i);
    const float4 b = *(const float4*)(src + i + 4);
    uint4 o;
    o.x = (unsigned)f2bf(a.x) | ((unsigned)f2bf(a.y) << 16);
    o.y = (unsigned)f2bf(a.z) | ((unsigned)f2bf(a.w) << 16);
    o.z = (unsigned)f2bf(b.x) | ((unsigned)f2bf(b.y) << 16);
    o.w = (unsigned)f2bf(b.z) | ((unsigned)f2bf(b.w) << 16);
    *(uint4*)(dst + i) = o;
}

// ---------------------------------------------------------------------------
// Kernel 1: router logits (fp32-exact), per-token softmax stats, zero fanout.
// ---------------------------------------------------------------------------
__global__ __launch_bounds__(256) void router_kernel(
    const float* __restrict__ x,        // [T, H]
    const float* __restrict__ rw,       // [E, H]
    float* __restrict__ logitsT,        // [E, T]
    float* __restrict__ maxv,           // [T]
    float* __restrict__ denom,          // [T]
    float* __restrict__ fanout)         // [T] (zeroed here)
{
    __shared__ float xs[4 * HDIM];
    __shared__ float sl[4 * NEXP];

    const int t0  = blockIdx.x * 4;
    const int tid = threadIdx.x;

    for (int i = tid * 4; i < 4 * HDIM; i += 256 * 4) {
        *(float4*)&xs[i] = *(const float4*)&x[(size_t)t0 * HDIM + i];
    }
    __syncthreads();

    const int e  = tid & 63;
    const int ts = tid >> 6;

    const float4* w4 = (const float4*)(rw + (size_t)e * HDIM);
    const float*  xr = xs + ts * HDIM;

    float acc = 0.0f;
    #pragma unroll 8
    for (int i = 0; i < HDIM / 4; ++i) {
        float4 wv = w4[i];
        acc = fmaf(xr[4 * i + 0], wv.x, acc);
        acc = fmaf(xr[4 * i + 1], wv.y, acc);
        acc = fmaf(xr[4 * i + 2], wv.z, acc);
        acc = fmaf(xr[4 * i + 3], wv.w, acc);
    }

    logitsT[(size_t)e * T_TOK + (t0 + ts)] = acc;
    sl[ts * NEXP + e] = acc;
    __syncthreads();

    if (tid < 4) {
        float mx = -INFINITY;
        #pragma unroll 8
        for (int i = 0; i < NEXP; ++i) mx = fmaxf(mx, sl[tid * NEXP + i]);
        float s = 0.0f;
        #pragma unroll 8
        for (int i = 0; i < NEXP; ++i) s += expf(sl[tid * NEXP + i] - mx);
        maxv[t0 + tid]   = mx;
        denom[t0 + tid]  = s;
        fanout[t0 + tid] = 0.0f;
    }
}

// ---------------------------------------------------------------------------
// Kernel 2: per-expert exact top-2048 (descending, ties->lower index) by
// bitonic-sorting 16384 packed keys in 128 KB of LDS.
// ---------------------------------------------------------------------------
__global__ __launch_bounds__(1024) void topk_sort_kernel(
    const float* __restrict__ logitsT,  // [E, T]
    const float* __restrict__ maxv,     // [T]
    const float* __restrict__ denom,    // [T]
    int*   __restrict__ idxArr,         // [E, CAP]
    float* __restrict__ out_li,         // [E*CAP]
    float* __restrict__ out_w,          // [E*CAP]
    float* __restrict__ out_fan)        // [T]
{
    extern __shared__ unsigned long long keys[];   // 16384 * 8B

    const int e   = blockIdx.x;
    const int tid = threadIdx.x;
    const float* lrow = logitsT + (size_t)e * T_TOK;

    for (int t = tid; t < T_TOK; t += 1024) {
        unsigned u = __float_as_uint(lrow[t]);
        u = (u & 0x80000000u) ? ~u : (u | 0x80000000u);
        keys[t] = ((unsigned long long)u << 32) | (unsigned)(~t);
    }
    __syncthreads();

    for (int k = 2; k <= T_TOK; k <<= 1) {
        for (int j = k >> 1; j > 0; j >>= 1) {
            for (int i = tid; i < T_TOK; i += 1024) {
                const int ixj = i ^ j;
                if (ixj > i) {
                    unsigned long long a = keys[i];
                    unsigned long long b = keys[ixj];
                    const bool doswap = ((i & k) == 0) ? (a < b) : (a > b);
                    if (doswap) { keys[i] = b; keys[ixj] = a; }
                }
            }
            __syncthreads();
        }
    }

    for (int c = tid; c < CAP; c += 1024) {
        const unsigned long long kk = keys[c];
        const int t = (int)(~(unsigned)kk);
        idxArr[e * CAP + c]  = t;
        out_li[e * CAP + c]  = (float)t;
        const float l = lrow[t];
        out_w[e * CAP + c]   = expf(l - maxv[t]) / denom[t];
        atomicAdd(&out_fan[t], 1.0f);
    }
}

// ---------------------------------------------------------------------------
// Expert FFN GEMM, all-bf16 operands, f32 accumulate (v_wmma_f32_16x16x32_bf16).
// A tile (gather-capable) staged with per-lane async global->LDS copies;
// B tile staged by the Tensor Data Mover (one tensor_load_to_lds per k-step,
// hardware-padded to the 80 B LDS row stride). Ping-pong buffers, one barrier
// per k-step, staging overlapped with WMMAs.
//   MODE 0: A = gather(xbf, idx)  epilogue relu(acc)^2 -> bf16 h
//   MODE 1: A = hbf               epilogue fp32 -> expert_out
// ---------------------------------------------------------------------------
template <int MODE>
__global__ __launch_bounds__(256) void ffn_gemm_kernel(
    const unsigned short* __restrict__ Abase,  // xbf [T,H] or hbf [E*CAP,D]
    const int* __restrict__ idxArr,            // [E, CAP]
    const unsigned short* __restrict__ Wbf,    // [E, 1024, 1024] bf16
    unsigned short* __restrict__ hOut,         // MODE 0 dest
    float* __restrict__ fOut)                  // MODE 1 dest
{
    constexpr int LDT = 40;   // ushort row stride: 80 B (16B-aligned, bank-clean)
    __shared__ __align__(16) unsigned short As[2][128 * LDT];
    __shared__ __align__(16) unsigned short Bs[2][128 * LDT];

    const int e     = blockIdx.z;
    const int mBase = blockIdx.y * 128;
    const int nBase = blockIdx.x * 128;
    const int tid   = threadIdx.x;
    const int lane  = tid & 31;
    const int wave  = tid >> 5;
    const int wm    = wave & 3;
    const int wn    = wave >> 2;
    const int half  = lane >> 4;
    const int l15   = lane & 15;

    // A staging map: 512 16B-chunks; thread covers (r0,c) and (r1,c).
    const int c  = tid & 3;        // k-chunk: elements [c*8, c*8+8)
    const int r0 = tid >> 2;       // rows 0..63
    const int r1 = r0 + 64;        // rows 64..127

    const unsigned short* gA0;
    const unsigned short* gA1;
    if (MODE == 0) {
        const int tok0 = idxArr[e * CAP + mBase + r0];
        const int tok1 = idxArr[e * CAP + mBase + r1];
        gA0 = Abase + (size_t)tok0 * HDIM;
        gA1 = Abase + (size_t)tok1 * HDIM;
    } else {
        gA0 = Abase + ((size_t)e * CAP + mBase + r0) * DDIM;
        gA1 = Abase + ((size_t)e * CAP + mBase + r1) * DDIM;
    }
    // B tile base (workgroup-uniform; consumed by the TDM descriptor).
    const unsigned short* gB = Wbf + ((size_t)e * 1024 + nBase) * 1024;

    v8f acc[2][4];
    const v8f vz = {0.f, 0.f, 0.f, 0.f, 0.f, 0.f, 0.f, 0.f};
    #pragma unroll
    for (int i = 0; i < 2; ++i)
        #pragma unroll
        for (int j = 0; j < 4; ++j) acc[i][j] = vz;

    const int cOff = c * 8;

    // Prologue: stage k-slab 0 into buffer 0.
    ASYNC_CP16(&As[0][r0 * LDT + cOff], gA0 + cOff);
    ASYNC_CP16(&As[0][r1 * LDT + cOff], gA1 + cOff);
    if (wave == 0) {
        tdm_load_tile_bf16((unsigned)(uintptr_t)&Bs[0][0], gB);
    }
    WAIT_ASYNC0();
    __builtin_amdgcn_s_wait_tensorcnt(0);
    __syncthreads();

    for (int it = 0; it < 1024 / 32; ++it) {
        const int cur = it & 1;
        const int nxt = cur ^ 1;

        // Stage next k-slab (overlaps with the WMMAs below).
        if (it + 1 < 1024 / 32) {
            const int k1 = (it + 1) * 32;
            ASYNC_CP16(&As[nxt][r0 * LDT + cOff], gA0 + k1 + cOff);
            ASYNC_CP16(&As[nxt][r1 * LDT + cOff], gA1 + k1 + cOff);
            if (wave == 0) {
                tdm_load_tile_bf16((unsigned)(uintptr_t)&Bs[nxt][0], gB + k1);
            }
        }

        // Fragment loads (ISA VGPR layouts) from the current buffer.
        V16U a[2], b[4];
        #pragma unroll
        for (int sm = 0; sm < 2; ++sm) {
            const unsigned short* p = &As[cur][(wm * 32 + sm * 16 + l15) * LDT];
            a[sm].h[0] = *(const v8bf*)(p + half * 8);      // K(j)=g*16+half*8+(j&7)
            a[sm].h[1] = *(const v8bf*)(p + 16 + half * 8);
        }
        #pragma unroll
        for (int sn = 0; sn < 4; ++sn) {
            const unsigned short* p = &Bs[cur][(wn * 64 + sn * 16 + l15) * LDT];
            b[sn].h[0] = *(const v8bf*)(p + half * 8);      // K(j)=g*16+half*8+(j&7)
            b[sn].h[1] = *(const v8bf*)(p + 16 + half * 8);
        }
        #pragma unroll
        for (int sm = 0; sm < 2; ++sm) {
            #pragma unroll
            for (int sn = 0; sn < 4; ++sn) {
                acc[sm][sn] = __builtin_amdgcn_wmma_f32_16x16x32_bf16(
                    false, a[sm].v, false, b[sn].v,
                    (short)0, acc[sm][sn], false, false);
            }
        }

        WAIT_ASYNC0();                         // this wave's A copies done
        __builtin_amdgcn_s_wait_tensorcnt(0);  // wave0: B tile done
        __syncthreads();                       // all writes visible; cur reusable
    }

    // Epilogue.
    #pragma unroll
    for (int sm = 0; sm < 2; ++sm) {
        #pragma unroll
        for (int sn = 0; sn < 4; ++sn) {
            const int n = nBase + wn * 64 + sn * 16 + l15;
            #pragma unroll
            for (int r = 0; r < 8; ++r) {
                const int m = mBase + wm * 32 + sm * 16 + half * 8 + r;
                float v = acc[sm][sn][r];
                const size_t off = ((size_t)e * CAP + m) * 1024 + n;
                if (MODE == 0) {
                    v = fmaxf(v, 0.0f);
                    v = v * v;                       // relu^2
                    hOut[off] = f2bf(v);
                } else {
                    fOut[off] = v;
                }
            }
        }
    }
}

// ---------------------------------------------------------------------------
// Launcher.
// d_ws: logitsT [E*T] f32 | maxv [T] | denom [T] | idx [E*CAP] i32 |
//       h [E*CAP*D] bf16 | xbf [T*H] bf16 | w1bf, w2bf [E*1024*1024] bf16
// d_out: expert_out [E*CAP*H] | local_indices [E*CAP] | weights [E*CAP] |
//        fanout [T]   (all f32)
// ---------------------------------------------------------------------------
extern "C" void kernel_launch(void* const* d_in, const int* in_sizes, int n_in,
                              void* d_out, int out_size, void* d_ws, size_t ws_size,
                              hipStream_t stream) {
    const float* x  = (const float*)d_in[0];
    const float* rw = (const float*)d_in[1];
    const float* w1 = (const float*)d_in[2];
    const float* w2 = (const float*)d_in[3];
    (void)in_sizes; (void)n_in; (void)out_size; (void)ws_size;

    float* out      = (float*)d_out;
    float* out_eo   = out;
    float* out_li   = out + (size_t)NEXP * CAP * HDIM;
    float* out_w    = out_li + (size_t)NEXP * CAP;
    float* out_fan  = out_w + (size_t)NEXP * CAP;

    float* logitsT  = (float*)d_ws;
    float* maxv     = logitsT + (size_t)NEXP * T_TOK;
    float* denom    = maxv + T_TOK;
    int*   idxArr   = (int*)(denom + T_TOK);
    unsigned short* hbf  = (unsigned short*)(idxArr + (size_t)NEXP * CAP);
    unsigned short* xbf  = hbf + (size_t)NEXP * CAP * DDIM;
    unsigned short* w1bf = xbf + (size_t)T_TOK * HDIM;
    unsigned short* w2bf = w1bf + (size_t)NEXP * DDIM * HDIM;

    const size_t nX = (size_t)T_TOK * HDIM;          // 16,777,216
    const size_t nW = (size_t)NEXP * DDIM * HDIM;    // 67,108,864

    // 0) one-shot bf16 conversions (RNE) of x, w1, w2
    cvt_bf16_kernel<<<(unsigned)(nX / 2048), 256, 0, stream>>>(x, xbf);
    cvt_bf16_kernel<<<(unsigned)(nW / 2048), 256, 0, stream>>>(w1, w1bf);
    cvt_bf16_kernel<<<(unsigned)(nW / 2048), 256, 0, stream>>>(w2, w2bf);

    // 1) router logits + softmax stats + fanout zero
    router_kernel<<<T_TOK / 4, 256, 0, stream>>>(x, rw, logitsT, maxv, denom, out_fan);

    // 2) per-expert exact top-2048 (bitonic sort, 128 KB dynamic LDS)
    topk_sort_kernel<<<NEXP, 1024, T_TOK * sizeof(unsigned long long), stream>>>(
        logitsT, maxv, denom, idxArr, out_li, out_w, out_fan);

    // 3) h = relu(gather(x) @ w1^T)^2 -> bf16 workspace
    dim3 gemmGrid(HDIM / 128, CAP / 128, NEXP);   // (8, 16, 64)
    ffn_gemm_kernel<0><<<gemmGrid, 256, 0, stream>>>(xbf, idxArr, w1bf, hbf, nullptr);

    // 4) expert_out = h @ w2^T -> fp32 output
    ffn_gemm_kernel<1><<<gemmGrid, 256, 0, stream>>>(hbf, idxArr, w2bf, nullptr, out_eo);
}
